// GatedDeltaMemoryBlock_87883620811457
// MI455X (gfx1250) — matrix operand order repaired and checked by reading, not compiled, not run on hardware
//
#include <hip/hip_runtime.h>
#include <cstdint>

// ---------------------------------------------------------------------------
// GatedDeltaMemoryBlock for MI455X (gfx1250, wave32, WMMA 16x16x32 f16).
//
//   2x mixer layer:  rms -> QKV gemm(wmma,f16) -> attn(64x64; Q/K tiles staged
//                    to LDS by the Tensor Data Mover, wmma from LDS)
//                    -> O gemm + residual -> rms -> MLP gemm (gelu fused)
//                    -> gemm + residual
//   GDN:             rms(+transpose to [B*N,T,D]) -> q/k/v/gate/beta/g proj
//                    -> causal conv4 + silu -> l2norm(q,k)
//                    -> delta-rule scan, S[128x256] f32 resident in LDS
//                       (131KB of the 320KB WGP LDS), fused gated RMSNorm
//                    -> output gemm (wmma) -> +residual, transpose back.
//
// GEMM: each wave computes a 16x64 strip (4 accumulators) so one A-fragment
// feeds 4 v_wmma per k-step; weights are preconverted to f16 K-major so
// B-fragments are two 16B loads per lane matching the ISA lane layout.
// ---------------------------------------------------------------------------

#define Dd   512
#define HM   8      // mixer heads
#define DH   64     // mixer head dim
#define GH   4      // gdn heads
#define GDK  128
#define GDV  256
#define Bb   2
#define TT   256
#define NN   64
#define NT   (Bb*TT*NN)   // 32768 tokens
#define MSEQ (Bb*NN)      // 128 gdn lanes
#define KEYD 512
#define VALD 1024

typedef __attribute__((ext_vector_type(16))) _Float16 v16h;
typedef __attribute__((ext_vector_type(8)))  _Float16 v8h;
typedef __attribute__((ext_vector_type(8)))  float    v8f;

#ifndef __has_builtin
#define __has_builtin(x) 0
#endif
#if __has_builtin(__builtin_amdgcn_tensor_load_to_lds) && \
    __has_builtin(__builtin_amdgcn_s_wait_tensorcnt)
#define GDM_HAVE_TDM 1
#else
#define GDM_HAVE_TDM 0
#endif

// ----------------------------- WMMA helpers --------------------------------

__device__ __forceinline__ v8f wmma_f16(v16h a, v16h b, v8f c) {
  // v_wmma_f32_16x16x32_f16  D = A*B + C
  return __builtin_amdgcn_wmma_f32_16x16x32_f16(
      /*neg_a=*/false, a, /*neg_b=*/false, b,
      /*c_mod=*/(short)0, c, /*reuse_a=*/false, /*reuse_b=*/false);
}

// A fragment: 16x32 f16, row-major source, lda in halves.
// lane m = lane&15 ; halves K = {(lane>>4)*8 .. +7} and {16+(lane>>4)*8 .. +7}
__device__ __forceinline__ v16h frag_a(const _Float16* row0, int lda, int k0, int lane) {
  const _Float16* p = row0 + (size_t)(lane & 15) * lda + k0 + ((lane >> 4) << 3);
  v8h lo = *reinterpret_cast<const v8h*>(p);
  v8h hi = *reinterpret_cast<const v8h*>(p + 16);
  v16h r;
#pragma unroll
  for (int i = 0; i < 8; ++i) { r[i] = lo[i]; r[i + 8] = hi[i]; }
  return r;
}

// B fragment: 32x16 f16 from K-major ("transposed") storage Bt[n][k], ldb in halves.
// lane n = lane&15 ; halves K = {(lane>>4)*8 .. +7} and {16+(lane>>4)*8 .. +7}
__device__ __forceinline__ v16h frag_b(const _Float16* col0, int ldb, int k0, int lane) {
  const _Float16* p = col0 + (size_t)(lane & 15) * ldb + k0 + ((lane >> 4) << 3);
  v8h lo = *reinterpret_cast<const v8h*>(p);
  v8h hi = *reinterpret_cast<const v8h*>(p + 16);
  v16h r;
#pragma unroll
  for (int i = 0; i < 8; ++i) { r[i] = lo[i]; r[i + 8] = hi[i]; }
  return r;
}

// --------------------- Tensor Data Mover tile staging ----------------------
#if GDM_HAVE_TDM
typedef unsigned int u32x4 __attribute__((ext_vector_type(4)));
typedef int          i32x4 __attribute__((ext_vector_type(4)));
typedef int          i32x8 __attribute__((ext_vector_type(8)));

__device__ __forceinline__ unsigned lds_byte_off(const void* p) {
  // generic LDS addresses carry the LDS byte offset in addr[31:0]
  return (unsigned)(uintptr_t)p;
}

// 2-D f16 tile load (tile_h x tile_w elements, row stride in elements) via TDM.
// D# bit layout per CDNA5 ISA section 8.3/8.4 (2-D tensor, groups 2/3 zero).
__device__ __forceinline__ void tdm_load_tile_f16(unsigned lds_addr,
                                                  const void* gsrc,
                                                  unsigned tile_w, unsigned tile_h,
                                                  unsigned stride_elems) {
  unsigned long long ga = (unsigned long long)(uintptr_t)gsrc;
  u32x4 g0;
  g0[0] = 1u;                                          // count=1 (valid user D#)
  g0[1] = lds_addr;                                    // lds_addr (bytes)
  g0[2] = (unsigned)(ga & 0xffffffffu);                // global_addr[31:0]
  g0[3] = (unsigned)((ga >> 32) & 0x01ffffffu) |       // global_addr[56:32]
          (2u << 30);                                  // type = 2 ("image")
  i32x8 g1;
  g1[0] = (int)(1u << 16);                             // wg_mask=0, data_size=1 (2B)
  g1[1] = (int)((stride_elems & 0xffffu) << 16);       // tensor_dim0 lo16
  g1[2] = (int)(((stride_elems >> 16) & 0xffffu) |     // tensor_dim0 hi16
                ((tile_h & 0xffffu) << 16));           // tensor_dim1 lo16
  g1[3] = (int)(((tile_h >> 16) & 0xffffu) |           // tensor_dim1 hi16
                ((tile_w & 0xffffu) << 16));           // tile_dim0
  g1[4] = (int)(tile_h & 0xffffu);                     // tile_dim1 (tile_dim2=0)
  g1[5] = (int)stride_elems;                           // tensor_dim0_stride lo32
  g1[6] = 0;                                           // stride hi | dim1_stride lo
  g1[7] = 0;
  i32x4 z4 = {0, 0, 0, 0};
#if __clang_major__ >= 23
  i32x8 z8 = {0, 0, 0, 0, 0, 0, 0, 0};
  __builtin_amdgcn_tensor_load_to_lds(g0, g1, z4, z4, z8, 0);
#else
  __builtin_amdgcn_tensor_load_to_lds(g0, g1, z4, z4, 0);
#endif
}
#endif  // GDM_HAVE_TDM

// ----------------------------- small kernels -------------------------------

// weight convert+transpose: Wt[n*K+k] = (f16)W[k*N+n]
__global__ void k_wt(const float* __restrict__ W, _Float16* __restrict__ Wt,
                     int K, int N) {
  long long idx = (long long)blockIdx.x * 256 + threadIdx.x;
  if (idx >= (long long)K * N) return;
  int k = (int)(idx % K);
  int n = (int)(idx / K);
  Wt[idx] = (_Float16)W[(size_t)k * N + n];
}

// RMSNorm + cast to f16. mode 0: row->row. mode 1: gather [B,T,N,D]->[B*N,T,D].
__global__ void __launch_bounds__(256) k_rms_cast(const float* __restrict__ X,
                                                  const float* __restrict__ w,
                                                  _Float16* __restrict__ Hout,
                                                  int mode) {
  int r = blockIdx.x;           // output row
  int t = threadIdx.x;
  int inRow = r;
  if (mode == 1) {
    int tt = r % TT; int s = r / TT; int b = s / NN; int n = s % NN;
    inRow = (b * TT + tt) * NN + n;
  }
  const float* x = X + (size_t)inRow * Dd;
  float a0 = x[t], a1 = x[t + 256];
  __shared__ float red[256];
  red[t] = a0 * a0 + a1 * a1;
  __syncthreads();
  for (int s = 128; s > 0; s >>= 1) { if (t < s) red[t] += red[t + s]; __syncthreads(); }
  float sc = rsqrtf(red[0] / (float)Dd + 1e-5f);
  _Float16* o = Hout + (size_t)r * Dd;
  o[t]       = (_Float16)(a0 * sc * w[t]);
  o[t + 256] = (_Float16)(a1 * sc * w[t + 256]);
}

// ----------------------------- WMMA GEMM -----------------------------------
// C[M,N] = epilogue(A[M,K]f16 * Bt[N,K]f16 (+Res))
// Each wave owns a 16x64 strip: one A-fragment feeds 4 WMMAs per k-step.
// EPI: 0=f32, 1=res+f32, 2=f16, 3=gelu->f16
template <int EPI>
__global__ void __launch_bounds__(256) k_gemm(const _Float16* __restrict__ A,
                                              const _Float16* __restrict__ Bt,
                                              const float* __restrict__ Res,
                                              float* __restrict__ Cf,
                                              _Float16* __restrict__ Ch,
                                              int M, int N, int K) {
  int lane = threadIdx.x & 31, wave = threadIdx.x >> 5;
  int strips_n = N >> 6;                       // 16x64 strips per tile-row
  long long strip = (long long)blockIdx.x * 8 + wave;
  int tm = (int)(strip / strips_n);
  int tn = (int)(strip - (long long)tm * strips_n);
  const _Float16* Arow = A + (size_t)tm * 16 * K;
  v8f acc[4];
#pragma unroll
  for (int j = 0; j < 4; ++j)
#pragma unroll
    for (int i = 0; i < 8; ++i) acc[j][i] = 0.f;

  const _Float16* apre = Arow + (size_t)(lane & 15) * K + ((lane >> 4) << 3);
  for (int k0 = 0; k0 < K; k0 += 32) {
    __builtin_prefetch(apre + k0 + 64, 0, 1);  // global_prefetch next A chunk
    v16h a = frag_a(Arow, K, k0, lane);
#pragma unroll
    for (int j = 0; j < 4; ++j) {
      v16h b = frag_b(Bt + (size_t)(tn * 64 + j * 16) * K, K, k0, lane);
      acc[j] = wmma_f16(a, b, acc[j]);
    }
  }

  int rbase = tm * 16 + ((lane >> 4) << 3);
  int cbase = tn * 64 + (lane & 15);
#pragma unroll
  for (int j = 0; j < 4; ++j) {
    int col = cbase + j * 16;
#pragma unroll
    for (int i = 0; i < 8; ++i) {
      size_t idx = (size_t)(rbase + i) * N + col;
      float u = acc[j][i];
      if (EPI == 0) {
        Cf[idx] = u;
      } else if (EPI == 1) {
        Cf[idx] = Res[idx] + u;
      } else if (EPI == 2) {
        Ch[idx] = (_Float16)u;
      } else {  // tanh-approx gelu
        float c = 0.7978845608028654f * (u + 0.044715f * u * u * u);
        Ch[idx] = (_Float16)(0.5f * u * (1.f + tanhf(c)));
      }
    }
  }
}

// --------------------------- mixer attention -------------------------------
// One block per (seq m, head h).  64x64 scores via WMMA from LDS tiles;
// Q/K tiles staged by the Tensor Data Mover when available.
__global__ void __launch_bounds__(256) k_attn(const _Float16* __restrict__ Q,
                                              const _Float16* __restrict__ K,
                                              const _Float16* __restrict__ V,
                                              _Float16* __restrict__ O) {
  __shared__ __attribute__((aligned(32))) _Float16 sQ[64 * 64];
  __shared__ __attribute__((aligned(32))) _Float16 sK[64 * 64];
  __shared__ __attribute__((aligned(32))) _Float16 sVt[64 * 64];
  __shared__ __attribute__((aligned(32))) _Float16 sP[64 * 64];
  __shared__ float sS[64 * 64];
  int m = blockIdx.x >> 3, h = blockIdx.x & 7;
  int tid = threadIdx.x, lane = tid & 31, wave = tid >> 5;
  size_t base = (size_t)m * 64 * Dd + (size_t)h * DH;

#if GDM_HAVE_TDM
  if (wave == 0) {   // TDM issues once per wave; EXEC ignored
    tdm_load_tile_f16(lds_byte_off(sQ), (const void*)(Q + base), 64, 64, Dd);
    tdm_load_tile_f16(lds_byte_off(sK), (const void*)(K + base), 64, 64, Dd);
  }
  for (int idx = tid; idx < 64 * 64; idx += 256) {   // V needs a transpose
    int n = idx >> 6, d = idx & 63;
    sVt[d * 64 + n] = V[base + (size_t)n * Dd + d];
  }
  __builtin_amdgcn_s_wait_tensorcnt(0);
  __syncthreads();
#else
  for (int idx = tid; idx < 64 * 64; idx += 256) {
    int n = idx >> 6, d = idx & 63;
    sQ[idx] = Q[base + (size_t)n * Dd + d];
    sK[idx] = K[base + (size_t)n * Dd + d];
    sVt[d * 64 + n] = V[base + (size_t)n * Dd + d];
  }
  __syncthreads();
#endif

  // S = Q K^T  (16 tiles / 8 waves)
  for (int tIdx = wave; tIdx < 16; tIdx += 8) {
    int tm = tIdx >> 2, tn = tIdx & 3;
    v8f acc = {0.f, 0.f, 0.f, 0.f, 0.f, 0.f, 0.f, 0.f};
#pragma unroll
    for (int k0 = 0; k0 < 64; k0 += 32) {
      v16h a = frag_a(sQ + (size_t)tm * 16 * 64, 64, k0, lane);
      v16h b = frag_b(sK + (size_t)tn * 16 * 64, 64, k0, lane);
      acc = wmma_f16(a, b, acc);
    }
    int col = tn * 16 + (lane & 15);
    int rbase = tm * 16 + ((lane >> 4) << 3);
#pragma unroll
    for (int i = 0; i < 8; ++i) sS[(rbase + i) * 64 + col] = acc[i];
  }
  __syncthreads();

  // softmax rows (scale 1/sqrt(64) = 0.125)
  if (tid < 64) {
    float mx = -1e30f;
    for (int j = 0; j < 64; ++j) mx = fmaxf(mx, sS[tid * 64 + j]);
    float sum = 0.f;
    for (int j = 0; j < 64; ++j) {
      float e = __expf((sS[tid * 64 + j] - mx) * 0.125f);
      sS[tid * 64 + j] = e; sum += e;
    }
    float inv = 1.f / sum;
    for (int j = 0; j < 64; ++j) sP[tid * 64 + j] = (_Float16)(sS[tid * 64 + j] * inv);
  }
  __syncthreads();

  // O = P V
  for (int tIdx = wave; tIdx < 16; tIdx += 8) {
    int tm = tIdx >> 2, tn = tIdx & 3;
    v8f acc = {0.f, 0.f, 0.f, 0.f, 0.f, 0.f, 0.f, 0.f};
#pragma unroll
    for (int k0 = 0; k0 < 64; k0 += 32) {
      v16h a = frag_a(sP + (size_t)tm * 16 * 64, 64, k0, lane);
      v16h b = frag_b(sVt + (size_t)tn * 16 * 64, 64, k0, lane);
      acc = wmma_f16(a, b, acc);
    }
    int col = tn * 16 + (lane & 15);
    int rbase = tm * 16 + ((lane >> 4) << 3);
#pragma unroll
    for (int i = 0; i < 8; ++i)
      O[base + (size_t)(rbase + i) * Dd + col] = (_Float16)acc[i];
  }
}

// ------------------------------ GDN pieces ---------------------------------

// beta = sigmoid(h@wb) ; g = -exp(A_log)*softplus(h@wa + dt_bias)
__global__ void __launch_bounds__(256) k_beta_g(const _Float16* __restrict__ Hl,
                                                const float* __restrict__ wb,
                                                const float* __restrict__ wa,
                                                const float* __restrict__ A_log,
                                                const float* __restrict__ dt_bias,
                                                float* __restrict__ Beta,
                                                float* __restrict__ Gl) {
  int r = blockIdx.x, t = threadIdx.x;
  int o = t >> 5, lane = t & 31;
  int h = o & 3; bool isA = (o >= 4);
  const float* W = isA ? wa : wb;
  const _Float16* x = Hl + (size_t)r * Dd;
  float acc = 0.f;
  for (int k = lane; k < Dd; k += 32) acc += (float)x[k] * W[k * GH + h];
  __shared__ float red[256];
  red[t] = acc;
  __syncthreads();
  if (lane == 0) {
    float s = 0.f;
    for (int j = 0; j < 32; ++j) s += red[o * 32 + j];
    if (!isA) {
      Beta[(size_t)r * GH + h] = 1.f / (1.f + __expf(-s));
    } else {
      float sp = s + dt_bias[h];
      sp = (sp > 20.f) ? sp : log1pf(__expf(sp));
      Gl[(size_t)r * GH + h] = -__expf(A_log[h]) * sp;
    }
  }
}

// depthwise causal conv (K=4) + silu
__global__ void k_conv_silu(const float* __restrict__ In, const float* __restrict__ Wc,
                            float* __restrict__ Out, int C) {
  long long i = (long long)blockIdx.x * 256 + threadIdx.x;
  int c = (int)(i % C);
  long long r = i / C;
  int t = (int)(r % TT);
  int m = (int)(r / TT);
  float acc = 0.f;
#pragma unroll
  for (int j = 0; j < 4; ++j) {
    int ts = t - 3 + j;
    if (ts >= 0) acc += In[((long long)(m * TT + ts)) * C + c] * Wc[c * 4 + j];
  }
  Out[i] = acc / (1.f + __expf(-acc));
}

// l2-normalize contiguous 128-vectors in place
__global__ void __launch_bounds__(128) k_l2norm(float* __restrict__ X) {
  int r = blockIdx.x, t = threadIdx.x;
  float v = X[(size_t)r * 128 + t];
  __shared__ float red[128];
  red[t] = v * v;
  __syncthreads();
  for (int s = 64; s > 0; s >>= 1) { if (t < s) red[t] += red[t + s]; __syncthreads(); }
  X[(size_t)r * 128 + t] = v * rsqrtf(red[0] + 1e-6f);
}

// Delta-rule scan. One block per (lane m, head h); S[128][256] f32 in LDS.
// Fused gated per-head RMSNorm epilogue -> f16 output.
__global__ void __launch_bounds__(256) k_scan(const float* __restrict__ Qc,
                                              const float* __restrict__ Kc,
                                              const float* __restrict__ Vc,
                                              const float* __restrict__ Beta,
                                              const float* __restrict__ Gl,
                                              const _Float16* __restrict__ Gate,
                                              const float* __restrict__ normw,
                                              _Float16* __restrict__ OV) {
  extern __shared__ float smem[];
  float* S   = smem;                // 128*256
  float* kv  = S + GDK * GDV;       // 128
  float* qv  = kv + GDK;            // 128
  float* red = qv + GDK;            // 8 wave partials
  int m = blockIdx.x >> 2, h = blockIdx.x & 3;
  int v = threadIdx.x;              // state column, 0..255
  int lane = v & 31, wave = v >> 5;

  for (int k = 0; k < GDK; ++k) S[k * GDV + v] = 0.f;
  __syncthreads();

  const float scale = 0.08838834764831845f;  // 1/sqrt(128)
  for (int t = 0; t < TT; ++t) {
    int row = m * TT + t;
    if (v < GDK) {
      kv[v] = Kc[(size_t)row * KEYD + h * GDK + v];
      qv[v] = Qc[(size_t)row * KEYD + h * GDK + v];
    }
    float bt  = Beta[(size_t)row * GH + h];
    float dec = __expf(Gl[(size_t)row * GH + h]);
    __syncthreads();

    float acc = 0.f;
#pragma unroll 8
    for (int k = 0; k < GDK; ++k) {
      float s = S[k * GDV + v] * dec;
      S[k * GDV + v] = s;
      acc += kv[k] * s;
    }
    float vt  = Vc[(size_t)row * VALD + h * GDV + v];
    float upd = bt * (vt - acc);
    float o = 0.f;
#pragma unroll 8
    for (int k = 0; k < GDK; ++k) {
      float s = S[k * GDV + v] + kv[k] * upd;
      S[k * GDV + v] = s;
      o += qv[k] * s;
    }
    o *= scale;

    // gated RMSNorm over the 256-wide head vector:
    // intra-wave shuffle reduce + one LDS pass across the 8 waves
    float ss = o * o;
#pragma unroll
    for (int off = 16; off > 0; off >>= 1) ss += __shfl_xor(ss, off, 32);
    if (lane == 0) red[wave] = ss;
    __syncthreads();
    float tot = 0.f;
#pragma unroll
    for (int j = 0; j < 8; ++j) tot += red[j];
    float rms = rsqrtf(tot / (float)GDV + 1e-5f);
    float gv = (float)Gate[(size_t)row * VALD + h * GDV + v];
    float outv = o * rms * normw[v] * (gv / (1.f + __expf(-gv)));
    OV[(size_t)row * VALD + h * GDV + v] = (_Float16)outv;
    __syncthreads();
  }
}

// out[b,t,n,:] = residual + y[(b*N+n)*T+t, :]
__global__ void k_final(const float* __restrict__ Y, const float* __restrict__ R,
                        float* __restrict__ Out) {
  size_t i = (size_t)blockIdx.x * 256 + threadIdx.x;
  int d = (int)(i & (Dd - 1));
  size_t row = i >> 9;
  int n = (int)(row % NN);
  size_t s = row / NN;
  int t = (int)(s % TT);
  int b = (int)(s / TT);
  size_t yrow = ((size_t)(b * NN + n)) * TT + t;
  Out[i] = R[i] + Y[yrow * Dd + d];
}

// ------------------------------- host side ---------------------------------

static const size_t MB = 1024ull * 1024ull;
// workspace layout (requires ws_size >= ~690 MB); regions reused across phases
#define OFF_X     (0 * MB)      // f32 [NT,512]  x / residual
#define OFF_H     (64 * MB)     // f16 [NT,512]  normalized activations
#define OFF_Q     (96 * MB)     // f16 [NT,512]
#define OFF_K     (128 * MB)    // f16 [NT,512]
#define OFF_V     (160 * MB)    // f16 [NT,512]
#define OFF_O     (192 * MB)    // f16 [NT,512]
#define OFF_MLP   (224 * MB)    // f16 [NT,2048]
#define OFF_W     (352 * MB)    // f16 transposed weights (~16 MB)
#define OFF_GATE  (368 * MB)    // f16 [NT,1024]
#define OFF_GQRAW (432 * MB)    // f32 [NT,512]
#define OFF_GKRAW (496 * MB)    // f32 [NT,512]
#define OFF_BETA  (560 * MB)    // f32 [NT,4]
#define OFF_GLOG  (561 * MB)    // f32 [NT,4]
#define OFF_QC    OFF_Q         // f32 [NT,512]   (reuses Q+K)
#define OFF_KC    OFF_V         // f32 [NT,512]   (reuses V+O)
#define OFF_VRAW  OFF_MLP       // f32 [NT,1024]  (reuses MLP)
#define OFF_VC    (562 * MB)    // f32 [NT,1024]
#define OFF_OV    OFF_GQRAW     // f16 [NT,1024]  (reuses GQRAW)
#define OFF_Y     OFF_MLP       // f32 [NT,512]   (reuses MLP after scan)

extern "C" void kernel_launch(void* const* d_in, const int* in_sizes, int n_in,
                              void* d_out, int out_size, void* d_ws, size_t ws_size,
                              hipStream_t stream) {
  (void)in_sizes; (void)n_in; (void)out_size; (void)ws_size;
  char* ws = (char*)d_ws;
  float*    X    = (float*)(ws + OFF_X);
  _Float16* Hh   = (_Float16*)(ws + OFF_H);
  _Float16* Qh   = (_Float16*)(ws + OFF_Q);
  _Float16* Kh   = (_Float16*)(ws + OFF_K);
  _Float16* Vh   = (_Float16*)(ws + OFF_V);
  _Float16* Oh   = (_Float16*)(ws + OFF_O);
  _Float16* MLPh = (_Float16*)(ws + OFF_MLP);
  _Float16* GATE = (_Float16*)(ws + OFF_GATE);
  float*    GQr  = (float*)(ws + OFF_GQRAW);
  float*    GKr  = (float*)(ws + OFF_GKRAW);
  float*    GVr  = (float*)(ws + OFF_VRAW);
  float*    QC   = (float*)(ws + OFF_QC);
  float*    KC   = (float*)(ws + OFF_KC);
  float*    VC   = (float*)(ws + OFF_VC);
  float*    BETA = (float*)(ws + OFF_BETA);
  float*    GLOG = (float*)(ws + OFF_GLOG);
  _Float16* OV   = (_Float16*)(ws + OFF_OV);
  float*    Y    = (float*)(ws + OFF_Y);

  // ---- inputs (setup_inputs flattening order) ----
  const float* state = (const float*)d_in[0];
  // mixer layer l base = 1 + 8*l : ln1,wq,wk,wv,wo,ln2,w1,w2
  const float* gdn_wq   = (const float*)d_in[17];
  const float* gdn_wk   = (const float*)d_in[18];
  const float* gdn_wv   = (const float*)d_in[19];
  const float* gdn_wb   = (const float*)d_in[20];
  const float* gdn_wa   = (const float*)d_in[21];
  const float* gdn_Alog = (const float*)d_in[22];
  const float* gdn_dtb  = (const float*)d_in[23];
  const float* conv_q   = (const float*)d_in[24];
  const float* conv_k   = (const float*)d_in[25];
  const float* conv_v   = (const float*)d_in[26];
  const float* gdn_wg   = (const float*)d_in[27];
  const float* gdn_nw   = (const float*)d_in[28];   // [256]
  const float* gdn_wo   = (const float*)d_in[29];
  const float* gdn_rmsw = (const float*)d_in[30];   // [512]

  // x := state
  hipMemcpyAsync(X, state, (size_t)NT * Dd * sizeof(float),
                 hipMemcpyDeviceToDevice, stream);

  // ---- convert weights to f16, K-major ----
  _Float16* Wbase = (_Float16*)(ws + OFF_W);
  size_t wcur = 0;
  auto nextW = [&](size_t halves) { _Float16* p = Wbase + wcur; wcur += halves; return p; };
  auto cvtW = [&](const float* W, _Float16* Wt, int K, int N) {
    long long total = (long long)K * N;
    k_wt<<<dim3((unsigned)((total + 255) / 256)), dim3(256), 0, stream>>>(W, Wt, K, N);
  };
  _Float16 *WQt[2], *WKt[2], *WVt[2], *WOt[2], *W1t[2], *W2t[2];
  for (int l = 0; l < 2; ++l) {
    int base = 1 + 8 * l;
    WQt[l] = nextW(512 * 512);  cvtW((const float*)d_in[base + 1], WQt[l], 512, 512);
    WKt[l] = nextW(512 * 512);  cvtW((const float*)d_in[base + 2], WKt[l], 512, 512);
    WVt[l] = nextW(512 * 512);  cvtW((const float*)d_in[base + 3], WVt[l], 512, 512);
    WOt[l] = nextW(512 * 512);  cvtW((const float*)d_in[base + 4], WOt[l], 512, 512);
    W1t[l] = nextW(512 * 2048); cvtW((const float*)d_in[base + 6], W1t[l], 512, 2048);
    W2t[l] = nextW(512 * 2048); cvtW((const float*)d_in[base + 7], W2t[l], 2048, 512);
  }
  _Float16* GWQt = nextW(512 * 512);   cvtW(gdn_wq, GWQt, 512, 512);
  _Float16* GWKt = nextW(512 * 512);   cvtW(gdn_wk, GWKt, 512, 512);
  _Float16* GWVt = nextW(512 * 1024);  cvtW(gdn_wv, GWVt, 512, 1024);
  _Float16* GWGt = nextW(512 * 1024);  cvtW(gdn_wg, GWGt, 512, 1024);
  _Float16* GWOt = nextW(1024 * 512);  cvtW(gdn_wo, GWOt, 1024, 512);

  auto gemm = [&](int epi, const _Float16* A, const _Float16* Bt, const float* Res,
                  float* Cf, _Float16* Ch, int M, int N, int K) {
    unsigned blocks = (unsigned)(((size_t)(M / 16) * (N / 64)) / 8);
    dim3 g(blocks), b(256);
    switch (epi) {
      case 0: k_gemm<0><<<g, b, 0, stream>>>(A, Bt, Res, Cf, Ch, M, N, K); break;
      case 1: k_gemm<1><<<g, b, 0, stream>>>(A, Bt, Res, Cf, Ch, M, N, K); break;
      case 2: k_gemm<2><<<g, b, 0, stream>>>(A, Bt, Res, Cf, Ch, M, N, K); break;
      default: k_gemm<3><<<g, b, 0, stream>>>(A, Bt, Res, Cf, Ch, M, N, K); break;
    }
  };

  // ---------------------------- mixer layers -------------------------------
  for (int l = 0; l < 2; ++l) {
    int base = 1 + 8 * l;
    const float* ln1 = (const float*)d_in[base + 0];
    const float* ln2 = (const float*)d_in[base + 5];
    k_rms_cast<<<dim3(NT), dim3(256), 0, stream>>>(X, ln1, Hh, 0);
    gemm(2, Hh, WQt[l], nullptr, nullptr, Qh, NT, 512, 512);
    gemm(2, Hh, WKt[l], nullptr, nullptr, Kh, NT, 512, 512);
    gemm(2, Hh, WVt[l], nullptr, nullptr, Vh, NT, 512, 512);
    k_attn<<<dim3(Bb * TT * HM), dim3(256), 0, stream>>>(Qh, Kh, Vh, Oh);
    gemm(1, Oh, WOt[l], X, X, nullptr, NT, 512, 512);     // x += o@wo
    k_rms_cast<<<dim3(NT), dim3(256), 0, stream>>>(X, ln2, Hh, 0);
    gemm(3, Hh, W1t[l], nullptr, nullptr, MLPh, NT, 2048, 512);  // gelu
    gemm(1, MLPh, W2t[l], X, X, nullptr, NT, 512, 2048);  // x += mlp@w2
  }

  // ------------------------------- GDN -------------------------------------
  // laned h = rms(state, gdn_rmsw), transposed to [B*N, T, D]
  k_rms_cast<<<dim3(NT), dim3(256), 0, stream>>>(X, gdn_rmsw, Hh, 1);
  gemm(0, Hh, GWQt, nullptr, GQr, nullptr, NT, 512, 512);
  gemm(0, Hh, GWKt, nullptr, GKr, nullptr, NT, 512, 512);
  gemm(0, Hh, GWVt, nullptr, GVr, nullptr, NT, 1024, 512);
  gemm(2, Hh, GWGt, nullptr, nullptr, GATE, NT, 1024, 512);
  k_beta_g<<<dim3(NT), dim3(256), 0, stream>>>(Hh, gdn_wb, gdn_wa, gdn_Alog,
                                               gdn_dtb, BETA, GLOG);
  k_conv_silu<<<dim3((NT * 512) / 256), dim3(256), 0, stream>>>(GQr, conv_q, QC, 512);
  k_conv_silu<<<dim3((NT * 512) / 256), dim3(256), 0, stream>>>(GKr, conv_k, KC, 512);
  k_conv_silu<<<dim3((NT * 1024) / 256), dim3(256), 0, stream>>>(GVr, conv_v, VC, 1024);
  k_l2norm<<<dim3(NT * GH), dim3(128), 0, stream>>>(QC);
  k_l2norm<<<dim3(NT * GH), dim3(128), 0, stream>>>(KC);

  size_t scanLds = (size_t)(GDK * GDV + GDK + GDK + 8) * sizeof(float); // ~132 KB
  k_scan<<<dim3(MSEQ * GH), dim3(256), scanLds, stream>>>(QC, KC, VC, BETA, GLOG,
                                                          GATE, gdn_nw, OV);

  gemm(0, OV, GWOt, nullptr, Y, nullptr, NT, 512, 1024);  // y = ov @ wo
  k_final<<<dim3((NT * Dd) / 256), dim3(256), 0, stream>>>(Y, X, (float*)d_out);
}
